// HGTEncoder_36352603193987
// MI455X (gfx1250) — compile-verified, matrix operand order checked by reading, not run on hardware
//
#include <hip/hip_runtime.h>

#define HID   256
#define HEADS 8
#define DDIM  32
#define LAYERS 3
#define INV_SQRT_D 0.17677669529663687f  // 1/sqrt(32)

typedef __attribute__((ext_vector_type(16))) __bf16 v16bf;
typedef __attribute__((ext_vector_type(8)))  __bf16 v8bf;
typedef __attribute__((ext_vector_type(8)))  float  v8f;

// gfx1250 async global->LDS DMA (ASYNCcnt-tracked), with graceful fallback.
// Probe (round 2 stderr) showed the builtin takes non-const AS1 int4* for the
// global address; LDS side is AS3.
#if defined(__has_builtin)
#  if __has_builtin(__builtin_amdgcn_global_load_async_to_lds_b128) && \
      __has_builtin(__builtin_amdgcn_s_wait_asynccnt)
#    define USE_ASYNC_LDS 1
#  endif
#endif
#ifndef USE_ASYNC_LDS
#  define USE_ASYNC_LDS 0
#endif

typedef int v4i_gcc __attribute__((vector_size(16)));
#define AS1 __attribute__((address_space(1)))
#define AS3 __attribute__((address_space(3)))

// ---------------- helpers ----------------

static __device__ __forceinline__ unsigned short f2bf(float f) {
  unsigned u = __float_as_uint(f);
  return (unsigned short)((u + 0x7FFFu + ((u >> 16) & 1u)) >> 16);  // RNE
}

// monotonic float -> uint key for atomicMax-based segment max
static __device__ __forceinline__ unsigned fenc(float f) {
  unsigned u = __float_as_uint(f);
  return (u & 0x80000000u) ? ~u : (u | 0x80000000u);
}
static __device__ __forceinline__ float fdec(unsigned k) {
  return (k & 0x80000000u) ? __uint_as_float(k & 0x7FFFFFFFu) : __uint_as_float(~k);
}

// Load one 16-element bf16 WMMA fragment for this lane.
// p0 is already offset by row*stride + khalf (khalf = 0 or 8 by lane>=16):
// frag[0..7]  = K base..base+7   at p0[0..7]
// frag[8..15] = K base+16..+23   at p0[16..23]
// matching the CDNA5 16-bit A/B lane layout.
static __device__ __forceinline__ v16bf load_frag(const unsigned short* p0) {
  v8bf lo = *(const v8bf*)(p0);
  v8bf hi = *(const v8bf*)(p0 + 16);
  return __builtin_shufflevector(lo, hi, 0,1,2,3,4,5,6,7,8,9,10,11,12,13,14,15);
}

// ---------------- GEMM: C[M,256] = A[M,256] @ B[256,256] (+ epilogue) --------
// A in bf16 row-major; Bt in bf16 as B-transposed ([f][c]) so fragment loads
// are contiguous. One block handles one 16-row M-tile: the A tile (16x256 bf16,
// 8KB) is staged once into LDS via async DMA and shared by all 8 waves; each
// wave computes one 16x16 tile (blockIdx.y*8+wave selects the N tile),
// K = 256 in 8 WMMA steps.
// mode 0: Cout = A@B + bias
// mode 1: out  = A@B + bias ; y = relu(s*out + (1-s)*xold) ; write f32 + bf16
__global__ void __launch_bounds__(256) gemm_wmma_bf16(
    const unsigned short* __restrict__ A, int M,
    const unsigned short* __restrict__ Bt,
    const float* __restrict__ bias,
    float* __restrict__ Cout, int mode,
    const float* __restrict__ xold,
    float* __restrict__ xnew,
    unsigned short* __restrict__ xnew_bf,
    const float* __restrict__ skipv, int layer)
{
  __shared__ unsigned short smemA[16 * HID];     // 8 KB A tile

  const int tid   = threadIdx.x;
  const int lane  = threadIdx.x & 31;
  const int wave  = threadIdx.x >> 5;
  const int row0  = blockIdx.x * 16;
  const int ntile = (blockIdx.y << 3) | wave;    // 0..15
  const int col0  = ntile * 16;

  // ---- stage A tile into LDS: 16 rows x 512B = 512 x 16B chunks ----
  {
    const int c0 = tid, c1 = tid + 256;
    const int r0 = c0 >> 5, o0 = (c0 & 31) << 3;   // element offsets (8 ushorts = 16B)
    const int r1 = c1 >> 5, o1 = (c1 & 31) << 3;
    int gr0 = row0 + r0; if (gr0 >= M) gr0 = M - 1;  // clamp; stores are guarded
    int gr1 = row0 + r1; if (gr1 >= M) gr1 = M - 1;
#if USE_ASYNC_LDS
    __builtin_amdgcn_global_load_async_to_lds_b128(
        (AS1 v4i_gcc*)(A + (size_t)gr0 * HID + o0),
        (AS3 v4i_gcc*)(smemA + r0 * HID + o0), 0, 0);
    __builtin_amdgcn_global_load_async_to_lds_b128(
        (AS1 v4i_gcc*)(A + (size_t)gr1 * HID + o1),
        (AS3 v4i_gcc*)(smemA + r1 * HID + o1), 0, 0);
    __builtin_amdgcn_s_wait_asynccnt(0);
#else
    *(uint4*)(smemA + r0 * HID + o0) = *(const uint4*)(A + (size_t)gr0 * HID + o0);
    *(uint4*)(smemA + r1 * HID + o1) = *(const uint4*)(A + (size_t)gr1 * HID + o1);
#endif
  }
  __syncthreads();

  const int bcol  = col0 + (lane & 15);
  const int khalf = (lane >> 4) << 3;            // 0 or 8

  const unsigned short* ap = smemA + (lane & 15) * HID + khalf;      // LDS
  const unsigned short* bp = Bt + (size_t)bcol * HID + khalf;        // global

  v8f acc = {0.f, 0.f, 0.f, 0.f, 0.f, 0.f, 0.f, 0.f};
#pragma unroll
  for (int kb = 0; kb < HID; kb += 32) {
    v16bf a = load_frag(ap + kb);
    v16bf b = load_frag(bp + kb);
    acc = __builtin_amdgcn_wmma_f32_16x16x32_bf16(
        false, a, false, b, (short)0, acc, false, false);
  }

  const int ccol  = col0 + (lane & 15);
  const int crow0 = row0 + ((lane >> 4) << 3);   // lanes 16-31 hold M=8..15
  const float bv  = bias[ccol];

  if (mode == 0) {
#pragma unroll
    for (int i = 0; i < 8; ++i) {
      int r = crow0 + i;
      if (r < M) Cout[(size_t)r * HID + ccol] = acc[i] + bv;
    }
  } else {
    const float s = 1.0f / (1.0f + __expf(-skipv[layer]));
#pragma unroll
    for (int i = 0; i < 8; ++i) {
      int r = crow0 + i;
      if (r < M) {
        float o = acc[i] + bv;
        float y = s * o + (1.0f - s) * xold[(size_t)r * HID + ccol];
        y = fmaxf(y, 0.0f);
        xnew[(size_t)r * HID + ccol]    = y;
        xnew_bf[(size_t)r * HID + ccol] = f2bf(y);
      }
    }
  }
}

// ---------------- weight preparation ----------------
// transpose+convert: Wt[f][c] = bf16(W[c][f]); bout = b
__global__ void prep_transpose(const float* __restrict__ W, const float* __restrict__ b,
                               unsigned short* __restrict__ Wt, float* __restrict__ bout)
{
  int f = blockIdx.x, c = threadIdx.x;
  Wt[f * HID + c] = f2bf(W[c * HID + f]);
  if (c == 0) bout[f] = b[f];
}

// fold per-head relation matrix R[h][d][e] into W (and bias):
//   Weff[c][h*32+e] = sum_d W[c][h*32+d] * R[h][d][e];  Wt = Weff^T in bf16
__global__ void prep_rel(const float* __restrict__ W, const float* __restrict__ b,
                         const float* __restrict__ R,
                         unsigned short* __restrict__ Wt, float* __restrict__ bout)
{
  int f = blockIdx.x, c = threadIdx.x;
  int h = f >> 5, e = f & 31;
  const float* Rc = R + h * DDIM * DDIM + e;     // stride DDIM over d
  const float* Wr = W + c * HID + h * DDIM;
  float acc = 0.f;
#pragma unroll
  for (int d = 0; d < DDIM; ++d) acc += Wr[d] * Rc[d * DDIM];
  Wt[f * HID + c] = f2bf(acc);
  if (c == 0) {
    float bb = 0.f;
    const float* br = b + h * DDIM;
#pragma unroll
    for (int d = 0; d < DDIM; ++d) bb += br[d] * Rc[d * DDIM];
    bout[f] = bb;
  }
}

// ---------------- elementwise ----------------
__global__ void fill_u32(unsigned* p, unsigned v, int n) {
  int i = blockIdx.x * blockDim.x + threadIdx.x;
  if (i < n) p[i] = v;
}
__global__ void fill_f32(float* p, float v, int n) {
  int i = blockIdx.x * blockDim.x + threadIdx.x;
  if (i < n) p[i] = v;
}
__global__ void cvt_f32_bf16(const float* __restrict__ in, unsigned short* __restrict__ out, int n) {
  int i = blockIdx.x * blockDim.x + threadIdx.x;
  if (i < n) out[i] = f2bf(in[i]);
}
__global__ void gelu_bf16(const float* __restrict__ in, unsigned short* __restrict__ out, int n) {
  int i = blockIdx.x * blockDim.x + threadIdx.x;
  if (i < n) {
    float x = in[i];
    float t = 0.7978845608028654f * (x + 0.044715f * x * x * x);
    out[i] = f2bf(0.5f * x * (1.0f + tanhf(t)));
  }
}

// ---------------- edge kernels ----------------
// one thread per (edge, head): logit = p[h]/sqrt(D) * <q[dst,h,:], kmsg[src,h,:]>
__global__ void edge_logits(const float* __restrict__ q, const float* __restrict__ kmsg,
                            const int* __restrict__ src, const int* __restrict__ dst,
                            const float* __restrict__ p, int E,
                            unsigned* __restrict__ segmax, float* __restrict__ logits)
{
  int t = blockIdx.x * blockDim.x + threadIdx.x;
  if (t >= E * HEADS) return;
  int e = t >> 3, h = t & 7;
  int s = src[e], d = dst[e];
  const float4* qp = (const float4*)(q    + (size_t)d * HID + h * DDIM);
  const float4* kp = (const float4*)(kmsg + (size_t)s * HID + h * DDIM);
  float acc = 0.f;
#pragma unroll
  for (int i = 0; i < 8; ++i) {
    float4 a = qp[i], b = kp[i];
    acc += a.x * b.x + a.y * b.y + a.z * b.z + a.w * b.w;
  }
  acc *= p[h] * INV_SQRT_D;
  logits[t] = acc;
  atomicMax(&segmax[d * HEADS + h], fenc(acc));
}

// exp(logit - segmax) in place, accumulate segment sums
__global__ void edge_expsum(const int* __restrict__ dst, const unsigned* __restrict__ segmax,
                            float* __restrict__ logits, float* __restrict__ segsum, int E)
{
  int t = blockIdx.x * blockDim.x + threadIdx.x;
  if (t >= E * HEADS) return;
  int e = t >> 3, h = t & 7;
  int d = dst[e];
  float ex = __expf(logits[t] - fdec(segmax[d * HEADS + h]));
  logits[t] = ex;
  atomicAdd(&segsum[d * HEADS + h], ex);
}

// one wave per edge: agg[dst] += vmsg[src] * alpha  (256 elems / 8 steps)
__global__ void edge_scatter(const int* __restrict__ src, const int* __restrict__ dst,
                             const float* __restrict__ vmsg, const float* __restrict__ logits,
                             const float* __restrict__ segsum, float* __restrict__ agg, int E)
{
  int gid = blockIdx.x * (blockDim.x >> 5) + (threadIdx.x >> 5);
  if (gid >= E) return;
  int lane = threadIdx.x & 31;
  int s = src[gid], d = dst[gid];
  const float* vrow = vmsg + (size_t)s * HID;
  float* arow = agg + (size_t)d * HID;
  __builtin_prefetch(vrow, 0, 0);              // global_prefetch_b8
#pragma unroll
  for (int h = 0; h < HEADS; ++h) {
    float wgt = logits[(size_t)gid * HEADS + h] / segsum[d * HEADS + h];
    int idx = h * 32 + lane;
    atomicAdd(&arow[idx], vrow[idx] * wgt);
  }
}

// ---------------- host orchestration ----------------
static inline int cdiv(int a, int b) { return (a + b - 1) / b; }

extern "C" void kernel_launch(void* const* d_in, const int* in_sizes, int n_in,
                              void* d_out, int out_size, void* d_ws, size_t ws_size,
                              hipStream_t stream)
{
  (void)n_in; (void)out_size; (void)ws_size;

  const float* x_drug  = (const float*)d_in[0];
  const float* x_react = (const float*)d_in[1];
  const int*   edge_dr = (const int*)d_in[2];
  const int*   edge_rd = (const int*)d_in[3];
  const float* Wkqv_d  = (const float*)d_in[4];
  const float* bkqv_d  = (const float*)d_in[5];
  const float* Wkqv_r  = (const float*)d_in[6];
  const float* bkqv_r  = (const float*)d_in[7];
  const float* oW_d    = (const float*)d_in[8];
  const float* ob_d    = (const float*)d_in[9];
  const float* oW_r    = (const float*)d_in[10];
  const float* ob_r    = (const float*)d_in[11];
  const float* skip_d  = (const float*)d_in[12];
  const float* skip_r  = (const float*)d_in[13];
  const float* a_dr    = (const float*)d_in[14];
  const float* m_dr    = (const float*)d_in[15];
  const float* p_dr    = (const float*)d_in[16];
  const float* a_rd    = (const float*)d_in[17];
  const float* m_rd    = (const float*)d_in[18];
  const float* p_rd    = (const float*)d_in[19];

  const int ND = in_sizes[0] / HID;
  const int NR = in_sizes[1] / HID;
  const int E  = in_sizes[2] / 2;
  const int* src_dr = edge_dr;  const int* dst_dr = edge_dr + E;
  const int* src_rd = edge_rd;  const int* dst_rd = edge_rd + E;

  // bump allocator over d_ws
  char* wptr = (char*)d_ws;
  auto alloc = [&](size_t bytes) -> void* {
    void* p = (void*)wptr;
    wptr += (bytes + 255) & ~(size_t)255;
    return p;
  };

  unsigned short* xbd  = (unsigned short*)alloc((size_t)ND * HID * 2);
  unsigned short* xbr  = (unsigned short*)alloc((size_t)NR * HID * 2);
  float* xdA = (float*)alloc((size_t)ND * HID * 4);
  float* xdB = (float*)alloc((size_t)ND * HID * 4);
  float* xrA = (float*)alloc((size_t)NR * HID * 4);
  float* xrB = (float*)alloc((size_t)NR * HID * 4);
  float* q_d    = (float*)alloc((size_t)ND * HID * 4);
  float* kmsg_d = (float*)alloc((size_t)ND * HID * 4);
  float* vmsg_d = (float*)alloc((size_t)ND * HID * 4);
  float* q_r    = (float*)alloc((size_t)NR * HID * 4);
  float* kmsg_r = (float*)alloc((size_t)NR * HID * 4);
  float* vmsg_r = (float*)alloc((size_t)NR * HID * 4);
  float* agg_d  = (float*)alloc((size_t)ND * HID * 4);
  float* agg_r  = (float*)alloc((size_t)NR * HID * 4);
  unsigned short* aggb = (unsigned short*)alloc((size_t)ND * HID * 2);
  float* logits_dr = (float*)alloc((size_t)E * HEADS * 4);
  float* logits_rd = (float*)alloc((size_t)E * HEADS * 4);
  unsigned* segmax_r = (unsigned*)alloc((size_t)NR * HEADS * 4);
  float*    segsum_r = (float*)   alloc((size_t)NR * HEADS * 4);
  unsigned* segmax_d = (unsigned*)alloc((size_t)ND * HEADS * 4);
  float*    segsum_d = (float*)   alloc((size_t)ND * HEADS * 4);
  unsigned short* Wt[8];
  float* bb[8];
  for (int i = 0; i < 8; ++i) {
    Wt[i] = (unsigned short*)alloc((size_t)HID * HID * 2);
    bb[i] = (float*)alloc(HID * 4);
  }
  // Wt/bb: 0=q_d 1=k_d 2=v_d 3=o_d 4=q_r 5=k_r 6=v_r 7=o_r

  auto gemm = [&](const unsigned short* A, int M, const unsigned short* Bt,
                  const float* bias, float* C, int mode, const float* xold,
                  float* xnew, unsigned short* xbf, const float* skipv, int l) {
    dim3 g(cdiv(M, 16), 2, 1);
    gemm_wmma_bf16<<<g, dim3(256), 0, stream>>>(A, M, Bt, bias, C, mode,
                                                xold, xnew, xbf, skipv, l);
  };

  // initial bf16 copies of x
  cvt_f32_bf16<<<cdiv(ND * HID, 256), 256, 0, stream>>>(x_drug, xbd, ND * HID);
  cvt_f32_bf16<<<cdiv(NR * HID, 256), 256, 0, stream>>>(x_react, xbr, NR * HID);

  const float* xd_old = x_drug;
  const float* xr_old = x_react;
  float* xd_ping[2] = {xdA, xdB};
  float* xr_ping[2] = {xrA, xrB};

  for (int l = 0; l < LAYERS; ++l) {
    const float* Wd = Wkqv_d + (size_t)l * 3 * HID * HID;
    const float* bd = bkqv_d + (size_t)l * 3 * HID;
    const float* Wr = Wkqv_r + (size_t)l * 3 * HID * HID;
    const float* br = bkqv_r + (size_t)l * 3 * HID;

    // weight prep (kqv: t=0 is k, t=1 is q, t=2 is v)
    prep_transpose<<<HID, HID, 0, stream>>>(Wd + 1 * HID * HID, bd + 1 * HID, Wt[0], bb[0]);
    prep_rel      <<<HID, HID, 0, stream>>>(Wd + 0 * HID * HID, bd + 0 * HID,
                                            a_dr + (size_t)l * HEADS * DDIM * DDIM, Wt[1], bb[1]);
    prep_rel      <<<HID, HID, 0, stream>>>(Wd + 2 * HID * HID, bd + 2 * HID,
                                            m_dr + (size_t)l * HEADS * DDIM * DDIM, Wt[2], bb[2]);
    prep_transpose<<<HID, HID, 0, stream>>>(oW_d + (size_t)l * HID * HID, ob_d + (size_t)l * HID, Wt[3], bb[3]);
    prep_transpose<<<HID, HID, 0, stream>>>(Wr + 1 * HID * HID, br + 1 * HID, Wt[4], bb[4]);
    prep_rel      <<<HID, HID, 0, stream>>>(Wr + 0 * HID * HID, br + 0 * HID,
                                            a_rd + (size_t)l * HEADS * DDIM * DDIM, Wt[5], bb[5]);
    prep_rel      <<<HID, HID, 0, stream>>>(Wr + 2 * HID * HID, br + 2 * HID,
                                            m_rd + (size_t)l * HEADS * DDIM * DDIM, Wt[6], bb[6]);
    prep_transpose<<<HID, HID, 0, stream>>>(oW_r + (size_t)l * HID * HID, ob_r + (size_t)l * HID, Wt[7], bb[7]);

    // node projections (relation matrices folded into k/v weights)
    gemm(xbd, ND, Wt[0], bb[0], q_d,    0, nullptr, nullptr, nullptr, nullptr, 0);
    gemm(xbd, ND, Wt[1], bb[1], kmsg_d, 0, nullptr, nullptr, nullptr, nullptr, 0);
    gemm(xbd, ND, Wt[2], bb[2], vmsg_d, 0, nullptr, nullptr, nullptr, nullptr, 0);
    gemm(xbr, NR, Wt[4], bb[4], q_r,    0, nullptr, nullptr, nullptr, nullptr, 0);
    gemm(xbr, NR, Wt[5], bb[5], kmsg_r, 0, nullptr, nullptr, nullptr, nullptr, 0);
    gemm(xbr, NR, Wt[6], bb[6], vmsg_r, 0, nullptr, nullptr, nullptr, nullptr, 0);

    // reset segment buffers
    fill_u32<<<cdiv(NR * HEADS, 256), 256, 0, stream>>>(segmax_r, 0u, NR * HEADS);
    fill_f32<<<cdiv(NR * HEADS, 256), 256, 0, stream>>>(segsum_r, 0.f, NR * HEADS);
    fill_f32<<<cdiv(NR * HID, 256), 256, 0, stream>>>(agg_r, 0.f, NR * HID);
    fill_u32<<<cdiv(ND * HEADS, 256), 256, 0, stream>>>(segmax_d, 0u, ND * HEADS);
    fill_f32<<<cdiv(ND * HEADS, 256), 256, 0, stream>>>(segsum_d, 0.f, ND * HEADS);
    fill_f32<<<cdiv(ND * HID, 256), 256, 0, stream>>>(agg_d, 0.f, ND * HID);

    // drug -> react relation
    int tEH = E * HEADS;
    edge_logits <<<cdiv(tEH, 256), 256, 0, stream>>>(q_r, kmsg_d, src_dr, dst_dr,
                                                     p_dr + (size_t)l * HEADS, E, segmax_r, logits_dr);
    edge_expsum <<<cdiv(tEH, 256), 256, 0, stream>>>(dst_dr, segmax_r, logits_dr, segsum_r, E);
    edge_scatter<<<cdiv(E, 8), 256, 0, stream>>>(src_dr, dst_dr, vmsg_d, logits_dr,
                                                 segsum_r, agg_r, E);
    // react -> drug relation
    edge_logits <<<cdiv(tEH, 256), 256, 0, stream>>>(q_d, kmsg_r, src_rd, dst_rd,
                                                     p_rd + (size_t)l * HEADS, E, segmax_d, logits_rd);
    edge_expsum <<<cdiv(tEH, 256), 256, 0, stream>>>(dst_rd, segmax_d, logits_rd, segsum_d, E);
    edge_scatter<<<cdiv(E, 8), 256, 0, stream>>>(src_rd, dst_rd, vmsg_r, logits_rd,
                                                 segsum_d, agg_d, E);

    // output projections + skip + relu (fused epilogue); last layer -> d_out
    float* xd_new = (l == LAYERS - 1) ? (float*)d_out : xd_ping[l & 1];
    float* xr_new = (l == LAYERS - 1) ? ((float*)d_out + (size_t)ND * HID) : xr_ping[l & 1];

    gelu_bf16<<<cdiv(ND * HID, 256), 256, 0, stream>>>(agg_d, aggb, ND * HID);
    gemm(aggb, ND, Wt[3], bb[3], nullptr, 1, xd_old, xd_new, xbd, skip_d, l);
    gelu_bf16<<<cdiv(NR * HID, 256), 256, 0, stream>>>(agg_r, aggb, NR * HID);
    gemm(aggb, NR, Wt[7], bb[7], nullptr, 1, xr_old, xr_new, xbr, skip_r, l);

    xd_old = xd_new;
    xr_old = xr_new;
  }
}
